// SIGNNLayer_15685220565559
// MI455X (gfx1250) — matrix operand order repaired and checked by reading, not compiled, not run on hardware
//
#include <hip/hip_runtime.h>
#include <stdint.h>

typedef __attribute__((ext_vector_type(16))) __bf16        v16bf;
typedef __attribute__((ext_vector_type(8)))  float         v8f;
typedef __attribute__((ext_vector_type(4)))  unsigned int  v4u;
typedef __attribute__((ext_vector_type(8)))  unsigned int  v8u;
typedef unsigned short u16;

#define DEV __device__ __forceinline__
// Same-wave LDS ops are in-order (DScnt in-order); only stop compiler reordering.
#define LDS_FENCE() asm volatile("" ::: "memory")

static const int NN = 100000;   // nodes
static const int NE = 1600000;  // edges

DEV u16 f2bf(float f) {
  unsigned u = __float_as_uint(f);
  u += 0x7fffu + ((u >> 16) & 1u);   // round-to-nearest-even
  return (u16)(u >> 16);
}
DEV unsigned pk2(float a, float b) {
  return (unsigned)f2bf(a) | ((unsigned)f2bf(b) << 16);
}

// A-fragment (16-bit A, 16xK): lane<16 -> K {o..o+7, o+16..o+23} with o=32*kk;
// lane>=16 -> shifted by 8.  rowp = LDS row of this lane's M (= lane&15).
DEV v16bf ldA(const u16* rowp, int kk, int half) {
  int o = 32 * kk + half * 8;
  v4u x = *(const v4u*)(rowp + o);
  v4u y = *(const v4u*)(rowp + o + 16);
  v8u z = __builtin_shufflevector(x, y, 0, 1, 2, 3, 4, 5, 6, 7);
  return __builtin_bit_cast(v16bf, z);
}
// B-fragment: pre-swizzled, 32 contiguous bytes per lane.
DEV v16bf ldB(const u16* p) {
  return __builtin_bit_cast(v16bf, *(const v8u*)p);
}
DEV v8f splat8(float x) { v8f r = {x, x, x, x, x, x, x, x}; return r; }
DEV v8f wmma_bf(v16bf a, v16bf b, v8f c) {
  return __builtin_amdgcn_wmma_f32_16x16x32_bf16(false, a, false, b, (short)0, c,
                                                 false, false);
}
DEV float sigm(float x) { return 1.0f / (1.0f + __expf(-x)); }

// Convert row-major W[K][ldN] into B-fragment layout [KK][NT][32][16] (bf16).
DEV void stageW(u16* dst, const float* W, int KK, int NT, int ldN,
                int tid, int nthreads) {
  int total = KK * NT * 32;
  for (int f = tid; f < total; f += nthreads) {
    int kk  = f / (NT * 32);
    int rem = f - kk * NT * 32;
    int nt  = rem >> 5;
    int l   = rem & 31;
    int col = nt * 16 + (l & 15);
    int k0  = kk * 32 + ((l < 16) ? 0 : 8);
    u16* o  = dst + (size_t)f * 16;
    #pragma unroll
    for (int j = 0; j < 16; j += 2)
      *(unsigned*)(o + j) =
          pk2(W[(size_t)(k0 + j + (j >= 8 ? 8 : 0)) * ldN + col], W[(size_t)(k0 + j + (j >= 8 ? 8 : 0) + 1) * ldN + col]);
  }
}
// Same, but the logical B is Wt^T with Wt stored [N][ldK] (e.g. torch w_ih).
DEV void stageWT(u16* dst, const float* Wt, int KK, int NT, int ldK,
                 int tid, int nthreads) {
  int total = KK * NT * 32;
  for (int f = tid; f < total; f += nthreads) {
    int kk  = f / (NT * 32);
    int rem = f - kk * NT * 32;
    int nt  = rem >> 5;
    int l   = rem & 31;
    int col = nt * 16 + (l & 15);
    int k0  = kk * 32 + ((l < 16) ? 0 : 8);
    const float* s = Wt + (size_t)col * ldK + k0;  // contiguous in k
    u16* o = dst + (size_t)f * 16;
    #pragma unroll
    for (int j = 0; j < 16; j += 2)
      *(unsigned*)(o + j) = pk2(s[j + (j >= 8 ? 8 : 0)], s[j + (j >= 8 ? 8 : 0) + 1]);
  }
}

__global__ void k_zero(float* p, int n) {
  int i = blockIdx.x * blockDim.x + threadIdx.x;
  if (i < n) p[i] = 0.0f;
}

// ---------------- message MLP + scatter-add -----------------------------
__global__ __launch_bounds__(256) void k_msg(
    const float* __restrict__ node_h, const float* __restrict__ edge_h,
    const float* __restrict__ mW1, const float* __restrict__ mb1,
    const float* __restrict__ mW2, const float* __restrict__ mb2,
    const int* __restrict__ eidx, float* __restrict__ agg) {
  __shared__ __attribute__((aligned(32))) u16 W1f[5 * 8 * 32 * 16];
  __shared__ __attribute__((aligned(32))) u16 W2f[4 * 4 * 32 * 16];
  __shared__ __attribute__((aligned(16))) u16 At[8][16 * 168];
  __shared__ __attribute__((aligned(16))) u16 Ht[8][16 * 136];
  __shared__ int Tg[8][16];

  stageW(W1f, mW1, 5, 8, 128, threadIdx.x, 256);
  stageW(W2f, mW2, 4, 4, 64, threadIdx.x, 256);
  __syncthreads();

  const int lane = threadIdx.x & 31;
  const int wave = threadIdx.x >> 5;
  const int l15  = lane & 15;
  const int half = lane >> 4;
  u16* myA = At[wave];
  u16* myH = Ht[wave];
  int* myT = Tg[wave];

  float bias1[8], bias2[4];
  #pragma unroll
  for (int nt = 0; nt < 8; ++nt) bias1[nt] = mb1[nt * 16 + l15];
  #pragma unroll
  for (int nt = 0; nt < 4; ++nt) bias2[nt] = mb2[nt * 16 + l15];

  const int ntiles = NE / 16;
  for (int tile = blockIdx.x * 8 + wave; tile < ntiles; tile += gridDim.x * 8) {
    int e = tile * 16 + l15;
    u16* arow = myA + l15 * 168;
    if (half == 0) {  // src node features -> cols [0,64)
      int s = eidx[e];
      const float4* p = (const float4*)(node_h + (size_t)s * 64);
      #pragma unroll
      for (int q = 0; q < 16; ++q) {
        float4 v = p[q];
        *(uint2*)(arow + q * 4) = make_uint2(pk2(v.x, v.y), pk2(v.z, v.w));
      }
    } else {  // tgt node -> [64,128), edge feats -> [128,160)
      int t = eidx[NE + e];
      myT[l15] = t;
      const float4* p = (const float4*)(node_h + (size_t)t * 64);
      #pragma unroll
      for (int q = 0; q < 16; ++q) {
        float4 v = p[q];
        *(uint2*)(arow + 64 + q * 4) = make_uint2(pk2(v.x, v.y), pk2(v.z, v.w));
      }
      const float4* pe = (const float4*)(edge_h + (size_t)e * 32);
      #pragma unroll
      for (int q = 0; q < 8; ++q) {
        float4 v = pe[q];
        *(uint2*)(arow + 128 + q * 4) = make_uint2(pk2(v.x, v.y), pk2(v.z, v.w));
      }
    }
    LDS_FENCE();

    // layer 1: [16x160] @ [160x128]
    v8f c[8];
    #pragma unroll
    for (int nt = 0; nt < 8; ++nt) c[nt] = splat8(bias1[nt]);
    #pragma unroll
    for (int kk = 0; kk < 5; ++kk) {
      v16bf a = ldA(arow, kk, half);
      #pragma unroll
      for (int nt = 0; nt < 8; ++nt)
        c[nt] = wmma_bf(a, ldB(W1f + ((kk * 8 + nt) * 32 + lane) * 16), c[nt]);
    }
    // ReLU -> bf16 -> LDS (C layout -> A layout)
    #pragma unroll
    for (int nt = 0; nt < 8; ++nt) {
      #pragma unroll
      for (int i = 0; i < 8; ++i)
        myH[(i + half * 8) * 136 + nt * 16 + l15] = f2bf(fmaxf(c[nt][i], 0.0f));
    }
    LDS_FENCE();

    // layer 2: [16x128] @ [128x64]
    v8f c2[4];
    #pragma unroll
    for (int nt = 0; nt < 4; ++nt) c2[nt] = splat8(bias2[nt]);
    const u16* hrow = myH + l15 * 136;
    #pragma unroll
    for (int kk = 0; kk < 4; ++kk) {
      v16bf a = ldA(hrow, kk, half);
      #pragma unroll
      for (int nt = 0; nt < 4; ++nt)
        c2[nt] = wmma_bf(a, ldB(W2f + ((kk * 4 + nt) * 32 + lane) * 16), c2[nt]);
    }
    // scatter-add messages into per-node aggregate
    #pragma unroll
    for (int nt = 0; nt < 4; ++nt) {
      int col = nt * 16 + l15;
      #pragma unroll
      for (int i = 0; i < 8; ++i) {
        int t = myT[i + half * 8];
        atomicAdd(agg + (size_t)t * 64 + col, c2[nt][i]);
      }
    }
    LDS_FENCE();
  }
}

// ---------------- GRU cell + LayerNorm ----------------------------------
__global__ __launch_bounds__(256) void k_gru(
    const float* __restrict__ node_h, const float* agg,
    const float* __restrict__ w_ih, const float* __restrict__ w_hh,
    const float* __restrict__ b_ih, const float* __restrict__ b_hh,
    const float* __restrict__ ln_g, const float* __restrict__ ln_b,
    float* outh) {
  __shared__ __attribute__((aligned(32))) u16 WI[2 * 12 * 32 * 16];
  __shared__ __attribute__((aligned(32))) u16 WH[2 * 12 * 32 * 16];
  __shared__ __attribute__((aligned(16))) u16 Ag[8][16 * 72];
  __shared__ __attribute__((aligned(16))) u16 Nh[8][16 * 72];

  stageWT(WI, w_ih, 2, 12, 64, threadIdx.x, 256);
  stageWT(WH, w_hh, 2, 12, 64, threadIdx.x, 256);
  __syncthreads();

  const int lane = threadIdx.x & 31;
  const int wave = threadIdx.x >> 5;
  const int l15  = lane & 15;
  const int half = lane >> 4;

  float brz[8], bin[4], bhn[4], lg[4], lb[4];
  #pragma unroll
  for (int nt = 0; nt < 8; ++nt) {
    int col = nt * 16 + l15;
    brz[nt] = b_ih[col] + b_hh[col];
  }
  #pragma unroll
  for (int j = 0; j < 4; ++j) {
    int col = 128 + j * 16 + l15;
    bin[j] = b_ih[col];
    bhn[j] = b_hh[col];
    lg[j]  = ln_g[j * 16 + l15];
    lb[j]  = ln_b[j * 16 + l15];
  }

  const int ntiles = NN / 16;  // 6250
  for (int tile = blockIdx.x * 8 + wave; tile < ntiles; tile += gridDim.x * 8) {
    int nb = tile * 16;
    // stage: half 0 -> aggregated, half 1 -> node_h (both bf16)
    {
      const float* sp = (half == 0 ? agg : node_h) + (size_t)(nb + l15) * 64;
      u16* dp = (half == 0 ? Ag[wave] : Nh[wave]) + l15 * 72;
      const float4* p = (const float4*)sp;
      #pragma unroll
      for (int q = 0; q < 16; ++q) {
        float4 v = p[q];
        *(uint2*)(dp + q * 4) = make_uint2(pk2(v.x, v.y), pk2(v.z, v.w));
      }
    }
    LDS_FENCE();

    v8f crz[8], cin[4], chn[4];
    #pragma unroll
    for (int nt = 0; nt < 8; ++nt) crz[nt] = splat8(brz[nt]);
    #pragma unroll
    for (int j = 0; j < 4; ++j) { cin[j] = splat8(bin[j]); chn[j] = splat8(bhn[j]); }

    const u16* aRow = Ag[wave] + l15 * 72;
    const u16* nRow = Nh[wave] + l15 * 72;
    #pragma unroll
    for (int kk = 0; kk < 2; ++kk) {
      v16bf aA = ldA(aRow, kk, half);
      v16bf aN = ldA(nRow, kk, half);
      #pragma unroll
      for (int nt = 0; nt < 8; ++nt) {  // r,z gates: gi + gh share accumulator
        crz[nt] = wmma_bf(aA, ldB(WI + ((kk * 12 + nt) * 32 + lane) * 16), crz[nt]);
        crz[nt] = wmma_bf(aN, ldB(WH + ((kk * 12 + nt) * 32 + lane) * 16), crz[nt]);
      }
      #pragma unroll
      for (int j = 0; j < 4; ++j) {     // n gate: keep gi/gh separate
        cin[j] = wmma_bf(aA, ldB(WI + ((kk * 12 + 8 + j) * 32 + lane) * 16), cin[j]);
        chn[j] = wmma_bf(aN, ldB(WH + ((kk * 12 + 8 + j) * 32 + lane) * 16), chn[j]);
      }
    }

    // gates + hidden update + LayerNorm (fp32 VALU, shfl reductions)
    #pragma unroll
    for (int i = 0; i < 8; ++i) {
      size_t nrow = (size_t)(nb + i + half * 8);
      float hv[4], s = 0.0f, q = 0.0f;
      #pragma unroll
      for (int j = 0; j < 4; ++j) {
        float rg = sigm(crz[j][i]);
        float zg = sigm(crz[4 + j][i]);
        float ng = tanhf(cin[j][i] + rg * chn[j][i]);
        float nh = node_h[nrow * 64 + j * 16 + l15];
        float h  = (1.0f - zg) * ng + zg * nh;
        hv[j] = h; s += h; q += h * h;
      }
      s += __shfl_xor(s, 1); s += __shfl_xor(s, 2);
      s += __shfl_xor(s, 4); s += __shfl_xor(s, 8);
      q += __shfl_xor(q, 1); q += __shfl_xor(q, 2);
      q += __shfl_xor(q, 4); q += __shfl_xor(q, 8);
      float mu = s * (1.0f / 64.0f);
      float var = q * (1.0f / 64.0f) - mu * mu;
      float rs = rsqrtf(var + 1e-5f);
      #pragma unroll
      for (int j = 0; j < 4; ++j)
        outh[nrow * 64 + j * 16 + l15] = (hv[j] - mu) * rs * lg[j] + lb[j];
    }
    LDS_FENCE();
  }
}

// ---------------- edge-update MLP (+residual, ReLU) ---------------------
__global__ __launch_bounds__(256) void k_edge(
    const float* __restrict__ hn, const float* __restrict__ edge_h,
    const float* __restrict__ eW1, const float* __restrict__ eb1,
    const float* __restrict__ eW2, const float* __restrict__ eb2,
    const int* __restrict__ eidx, float* __restrict__ oute) {
  __shared__ __attribute__((aligned(32))) u16 W1f[5 * 8 * 32 * 16];
  __shared__ __attribute__((aligned(32))) u16 W2f[4 * 2 * 32 * 16];
  __shared__ __attribute__((aligned(16))) u16 At[8][16 * 168];
  __shared__ __attribute__((aligned(16))) u16 Ht[8][16 * 136];

  stageW(W1f, eW1, 5, 8, 128, threadIdx.x, 256);
  stageW(W2f, eW2, 4, 2, 32, threadIdx.x, 256);
  __syncthreads();

  const int lane = threadIdx.x & 31;
  const int wave = threadIdx.x >> 5;
  const int l15  = lane & 15;
  const int half = lane >> 4;
  u16* myA = At[wave];
  u16* myH = Ht[wave];

  float bias1[8], bias2[2];
  #pragma unroll
  for (int nt = 0; nt < 8; ++nt) bias1[nt] = eb1[nt * 16 + l15];
  #pragma unroll
  for (int nt = 0; nt < 2; ++nt) bias2[nt] = eb2[nt * 16 + l15];

  const int ntiles = NE / 16;
  for (int tile = blockIdx.x * 8 + wave; tile < ntiles; tile += gridDim.x * 8) {
    int e = tile * 16 + l15;
    u16* arow = myA + l15 * 168;
    if (half == 0) {
      int s = eidx[e];
      const float4* p = (const float4*)(hn + (size_t)s * 64);
      #pragma unroll
      for (int q = 0; q < 16; ++q) {
        float4 v = p[q];
        *(uint2*)(arow + q * 4) = make_uint2(pk2(v.x, v.y), pk2(v.z, v.w));
      }
    } else {
      int t = eidx[NE + e];
      const float4* p = (const float4*)(hn + (size_t)t * 64);
      #pragma unroll
      for (int q = 0; q < 16; ++q) {
        float4 v = p[q];
        *(uint2*)(arow + 64 + q * 4) = make_uint2(pk2(v.x, v.y), pk2(v.z, v.w));
      }
      const float4* pe = (const float4*)(edge_h + (size_t)e * 32);
      #pragma unroll
      for (int q = 0; q < 8; ++q) {
        float4 v = pe[q];
        *(uint2*)(arow + 128 + q * 4) = make_uint2(pk2(v.x, v.y), pk2(v.z, v.w));
      }
    }
    LDS_FENCE();

    v8f c[8];
    #pragma unroll
    for (int nt = 0; nt < 8; ++nt) c[nt] = splat8(bias1[nt]);
    #pragma unroll
    for (int kk = 0; kk < 5; ++kk) {
      v16bf a = ldA(arow, kk, half);
      #pragma unroll
      for (int nt = 0; nt < 8; ++nt)
        c[nt] = wmma_bf(a, ldB(W1f + ((kk * 8 + nt) * 32 + lane) * 16), c[nt]);
    }
    #pragma unroll
    for (int nt = 0; nt < 8; ++nt) {
      #pragma unroll
      for (int i = 0; i < 8; ++i)
        myH[(i + half * 8) * 136 + nt * 16 + l15] = f2bf(fmaxf(c[nt][i], 0.0f));
    }
    LDS_FENCE();

    v8f c2[2];
    #pragma unroll
    for (int nt = 0; nt < 2; ++nt) c2[nt] = splat8(bias2[nt]);
    const u16* hrow = myH + l15 * 136;
    #pragma unroll
    for (int kk = 0; kk < 4; ++kk) {
      v16bf a = ldA(hrow, kk, half);
      #pragma unroll
      for (int nt = 0; nt < 2; ++nt)
        c2[nt] = wmma_bf(a, ldB(W2f + ((kk * 2 + nt) * 32 + lane) * 16), c2[nt]);
    }
    // residual + ReLU + store
    #pragma unroll
    for (int nt = 0; nt < 2; ++nt) {
      int col = nt * 16 + l15;
      #pragma unroll
      for (int i = 0; i < 8; ++i) {
        size_t e2 = (size_t)(tile * 16 + i + half * 8);
        float res = edge_h[e2 * 32 + col];
        oute[e2 * 32 + col] = fmaxf(c2[nt][i] + res, 0.0f);
      }
    }
    LDS_FENCE();
  }
}

extern "C" void kernel_launch(void* const* d_in, const int* in_sizes, int n_in,
                              void* d_out, int out_size, void* d_ws, size_t ws_size,
                              hipStream_t stream) {
  const float* node_h = (const float*)d_in[0];
  const float* edge_h = (const float*)d_in[1];
  const float* mW1 = (const float*)d_in[2];
  const float* mb1 = (const float*)d_in[3];
  const float* mW2 = (const float*)d_in[4];
  const float* mb2 = (const float*)d_in[5];
  const float* w_ih = (const float*)d_in[6];
  const float* w_hh = (const float*)d_in[7];
  const float* b_ih = (const float*)d_in[8];
  const float* b_hh = (const float*)d_in[9];
  const float* ln_g = (const float*)d_in[10];
  const float* ln_b = (const float*)d_in[11];
  const float* eW1 = (const float*)d_in[12];
  const float* eb1 = (const float*)d_in[13];
  const float* eW2 = (const float*)d_in[14];
  const float* eb2 = (const float*)d_in[15];
  const int* eidx = (const int*)d_in[16];

  float* outh = (float*)d_out;                      // [NN,64]; also aggregate scratch
  float* oute = (float*)d_out + (size_t)NN * 64;    // [NE,32]

  k_zero<<<(NN * 64 + 255) / 256, 256, 0, stream>>>(outh, NN * 64);
  k_msg<<<2048, 256, 0, stream>>>(node_h, edge_h, mW1, mb1, mW2, mb2, eidx, outh);
  k_gru<<<512, 256, 0, stream>>>(node_h, outh, w_ih, w_hh, b_ih, b_hh, ln_g, ln_b, outh);
  k_edge<<<2048, 256, 0, stream>>>(outh, edge_h, eW1, eb1, eW2, eb2, eidx, oute);
}
